// ConT_7730941133030
// MI455X (gfx1250) — matrix operand order repaired and checked by reading, not compiled, not run on hardware
//
#include <hip/hip_runtime.h>

// ---------------------------------------------------------------------------
// CDNA5 (gfx1250) implementation. wave32. Matrix ops via v_wmma_f32_16x16x32_bf16.
// Async global->LDS fills (ASYNCcnt) with double-buffered LDS when available.
//
// Key observation: the routing permutation (6 rounds of argsort) is applied and
// then exactly inverted around a purely per-token op, so it cancels:
//   out = x + (softmax_d((q-k)/8) * v) @ W_proj + b_proj
// ---------------------------------------------------------------------------

typedef __attribute__((ext_vector_type(16))) __bf16 v16bf;
typedef __attribute__((ext_vector_type(8)))  float  v8f;
typedef __attribute__((ext_vector_type(4)))  int    v4i;

#define LDT    40   // padded LDS row stride in bf16 elements (32 data + 8 pad)
#define TILE_K 32

#if defined(__has_builtin)
#if __has_builtin(__builtin_amdgcn_global_load_async_to_lds_b128)
#define HAS_ASYNC_LDS 1
#endif
#endif

#ifndef HAS_ASYNC_LDS
#define HAS_ASYNC_LDS 0
#endif

#if HAS_ASYNC_LDS
typedef __attribute__((address_space(1))) v4i gv4i;   // global
typedef __attribute__((address_space(3))) v4i lv4i;   // LDS
#if defined(__has_builtin) && __has_builtin(__builtin_amdgcn_s_wait_asynccnt)
#define WAIT_ASYNC(N) __builtin_amdgcn_s_wait_asynccnt(N)
#else
#define WAIT_ASYNC(N) asm volatile("s_wait_asynccnt %0" ::"i"(N) : "memory")
#endif
#endif

__device__ __forceinline__ unsigned bf16_rne(float x) {
    unsigned u = __float_as_uint(x);
    return (u + 0x7FFFu + ((u >> 16) & 1u)) >> 16;   // round-to-nearest-even
}

union AFrag { v16bf v; unsigned u[8]; };
union BFrag { v16bf v; uint4    q[2]; };

// A-matrix 16x32 bf16 fragment (ISA 7.12.2): lane = row%16 | half<<4,
// VGPR v halves cover K = half*8 + [0..7] and 16 + half*8 + [0..7].
__device__ __forceinline__ v16bf load_a_frag(const unsigned short* tile, int lane) {
    const int row = lane & 15, half = lane >> 4;
    const unsigned short* r = tile + row * LDT;
    AFrag f;
#pragma unroll
    for (int v = 0; v < 4; ++v)
        f.u[v]     = *(const unsigned*)(r + half * 8 + 2 * v);
#pragma unroll
    for (int v = 0; v < 4; ++v)
        f.u[4 + v] = *(const unsigned*)(r + 16 + half * 8 + 2 * v);
    return f.v;
}

// B-matrix 32x16 bf16 fragment: n = lane%16, k = (lane/16)*8 + e (e<8), 16 + (lane/16)*8 + (e-8) (e>=8).
// B stored column-major (B^T row-major) in LDS -> two ds_load_b128 per lane.
__device__ __forceinline__ v16bf load_b_frag(const unsigned short* tile, int lane) {
    const int n = lane & 15, half = lane >> 4;
    const unsigned short* r = tile + n * LDT + half * 8;
    BFrag f;
    f.q[0] = *(const uint4*)(r);
    f.q[1] = *(const uint4*)(r + 16);
    return f.v;
}

#define WMMA_BF16(A, B, C) \
    __builtin_amdgcn_wmma_f32_16x16x32_bf16(false, (A), false, (B), (short)0, (C), false, false)

// one K-step of the 32x32 wave tile: 4 fragment loads, 4 WMMAs
__device__ __forceinline__ void mma_step(const unsigned short* tA,
                                         const unsigned short* tB,
                                         int lane, int wm, int wn,
                                         v8f& c00, v8f& c01, v8f& c10, v8f& c11)
{
    v16bf a0 = load_a_frag(tA + (32 * wm)      * LDT, lane);
    v16bf a1 = load_a_frag(tA + (32 * wm + 16) * LDT, lane);
    v16bf b0 = load_b_frag(tB + (32 * wn)      * LDT, lane);
    v16bf b1 = load_b_frag(tB + (32 * wn + 16) * LDT, lane);
    c00 = WMMA_BF16(a0, b0, c00);
    c01 = WMMA_BF16(a0, b1, c01);
    c10 = WMMA_BF16(a1, b0, c10);
    c11 = WMMA_BF16(a1, b1, c11);
}

#if HAS_ASYNC_LDS
// Issue async global->LDS fills for one 128x32 A tile + 64x32 B tile.
// 3 async instructions per thread (= ASYNCcnt += 3 per wave).
__device__ __forceinline__ void fill_async(const unsigned short* A,
                                           const unsigned short* Bt,
                                           unsigned short* sA, unsigned short* sB,
                                           long mBase, long nBase, long K, int k0,
                                           int tid)
{
#pragma unroll
    for (int i = 0; i < 2; ++i) {
        const int c = tid + i * 256;           // 512 chunks of 8 bf16
        const int row = c >> 2, col = (c & 3) * 8;
        const unsigned short* g = A + (mBase + row) * K + k0 + col;
        unsigned short* l = sA + row * LDT + col;
        __builtin_amdgcn_global_load_async_to_lds_b128(
            (gv4i*)(unsigned short*)g, (lv4i*)l, 0, 0);
    }
    {
        const int row = tid >> 2, col = (tid & 3) * 8;   // 256 chunks
        const unsigned short* g = Bt + (nBase + row) * K + k0 + col;
        unsigned short* l = sB + row * LDT + col;
        __builtin_amdgcn_global_load_async_to_lds_b128(
            (gv4i*)(unsigned short*)g, (lv4i*)l, 0, 0);
    }
}
#endif

// ---------------------------------------------------------------------------
// GEMM: C[M][N] = A[M][K](bf16,row-major) * Bt[N][K](bf16, = B column-major)
//       + bias[N] (+ residual[M][N] if non-null), C fp32.
// Block: 256 thr (8 waves), block tile 128x64, wave tile 32x32, K-step 32.
// M multiple of 128, N multiple of 64, K multiple of 32 (true for all calls).
// ---------------------------------------------------------------------------
__global__ void __launch_bounds__(256)
gemm_bf16_wmma(const unsigned short* __restrict__ A,
               const unsigned short* __restrict__ Bt,
               const float* __restrict__ bias,
               const float* __restrict__ residual,
               float* __restrict__ C,
               int M, int N, int K)
{
    __shared__ __align__(16) unsigned short sA[2][128 * LDT];
    __shared__ __align__(16) unsigned short sB[2][64 * LDT];

    const int tid  = threadIdx.x;
    const int lane = tid & 31;
    const int wid  = tid >> 5;
    const int wm   = wid & 3;      // 0..3 -> 32-row strip
    const int wn   = wid >> 2;     // 0..1 -> 32-col strip

    const long mBase = (long)blockIdx.y * 128;
    const long nBase = (long)blockIdx.x * 64;
    const long Kl = K;

    v8f c00 = {}, c01 = {}, c10 = {}, c11 = {};
    const int nT = K / TILE_K;

#if HAS_ASYNC_LDS
    // ---- async double-buffered pipeline (CDNA5 GLOBAL_LOAD_ASYNC_TO_LDS) ----
    fill_async(A, Bt, sA[0], sB[0], mBase, nBase, Kl, 0, tid);
    for (int t = 0; t < nT - 1; ++t) {
        const int cur = t & 1, nxt = cur ^ 1;
        fill_async(A, Bt, sA[nxt], sB[nxt], mBase, nBase, Kl, (t + 1) * TILE_K, tid);
        WAIT_ASYNC(3);          // this wave's fills for buffer `cur` are done
        __syncthreads();        // everyone's fills for `cur` are done
        mma_step(sA[cur], sB[cur], lane, wm, wn, c00, c01, c10, c11);
        __syncthreads();        // reads of `cur` done before it is refilled
    }
    WAIT_ASYNC(0);
    __syncthreads();
    mma_step(sA[(nT - 1) & 1], sB[(nT - 1) & 1], lane, wm, wn, c00, c01, c10, c11);
#else
    // ---- fallback: register-prefetch double-buffered pipeline ----
    const int aRow = tid >> 2, aCol = (tid & 3) * 8;   // A: 2 chunks/thread
    const int bRow = tid >> 2, bCol = (tid & 3) * 8;   // B: 1 chunk/thread
    uint4 ra0 = *(const uint4*)(A + (mBase + aRow) * Kl + aCol);
    uint4 ra1 = *(const uint4*)(A + (mBase + aRow + 64) * Kl + aCol);
    uint4 rb  = *(const uint4*)(Bt + (nBase + bRow) * Kl + bCol);
    for (int t = 0; t < nT; ++t) {
        const int cur = t & 1;
        *(uint4*)(sA[cur] + aRow * LDT + aCol)        = ra0;
        *(uint4*)(sA[cur] + (aRow + 64) * LDT + aCol) = ra1;
        *(uint4*)(sB[cur] + bRow * LDT + bCol)        = rb;
        if (t + 1 < nT) {
            const int k0 = (t + 1) * TILE_K;
            ra0 = *(const uint4*)(A + (mBase + aRow) * Kl + k0 + aCol);
            ra1 = *(const uint4*)(A + (mBase + aRow + 64) * Kl + k0 + aCol);
            rb  = *(const uint4*)(Bt + (nBase + bRow) * Kl + k0 + bCol);
        }
        __syncthreads();
        mma_step(sA[cur], sB[cur], lane, wm, wn, c00, c01, c10, c11);
        __syncthreads();
    }
#endif

    // C/D layout: VGPR r -> M = r + 8*(lane/16), N = lane%16
    const int  half = lane >> 4;
    const long row0 = mBase + 32 * wm + 8 * half;
    const long col0 = nBase + 32 * wn + (lane & 15);
    const v8f* accs[4] = { &c00, &c01, &c10, &c11 };
#pragma unroll
    for (int ai = 0; ai < 2; ++ai) {
#pragma unroll
        for (int bi = 0; bi < 2; ++bi) {
            const v8f& a = *accs[ai * 2 + bi];
            const long col = col0 + 16 * bi;
            const float bv = bias[col];
#pragma unroll
            for (int r = 0; r < 8; ++r) {
                const long rr = row0 + 16 * ai + r;
                float o = a[r] + bv;
                if (residual) o += residual[rr * N + col];
                C[rr * N + col] = o;
            }
        }
    }
}

// ---------------------------------------------------------------------------
// Per-token head-dim softmax:  o = softmax((q-k)*0.125, over d=64) * v.
// One wave per (b,s,h); lane holds dims {2*lane, 2*lane+1}. Output packed bf16
// row-major [B*S][512] (column = h*64 + d), ready as GEMM2's A matrix.
// ---------------------------------------------------------------------------
__global__ void __launch_bounds__(256)
attn_softmax_mul(const float* __restrict__ qkv,   // [B*S][1536]  (q|k|v blocks)
                 unsigned* __restrict__ out)      // [B*S][256] uint = 2 bf16
{
    const int  lane = threadIdx.x & 31;
    const int  wid  = threadIdx.x >> 5;
    const long job  = (long)blockIdx.x * 8 + wid;  // B*S*H jobs total
    const long bs   = job >> 3;
    const int  h    = (int)(job & 7);

    const float* base = qkv + bs * 1536 + h * 64;
    const float2 q = *(const float2*)(base +        2 * lane);
    const float2 k = *(const float2*)(base +  512 + 2 * lane);
    const float2 v = *(const float2*)(base + 1024 + 2 * lane);

    const float d0 = (q.x - k.x) * 0.125f;
    const float d1 = (q.y - k.y) * 0.125f;

    float m = fmaxf(d0, d1);
#pragma unroll
    for (int off = 16; off > 0; off >>= 1)
        m = fmaxf(m, __shfl_xor(m, off, 32));

    const float e0 = __expf(d0 - m);
    const float e1 = __expf(d1 - m);
    float s = e0 + e1;
#pragma unroll
    for (int off = 16; off > 0; off >>= 1)
        s += __shfl_xor(s, off, 32);

    const float inv = 1.0f / s;
    const float o0 = e0 * inv * v.x;
    const float o1 = e1 * inv * v.y;

    out[bs * 256 + h * 32 + lane] = bf16_rne(o0) | (bf16_rne(o1) << 16);
}

// ---------------------------------------------------------------------------
// fp32 -> bf16 converters (plain, and transposing for weights: W[K][N] -> Wt[N][K])
// ---------------------------------------------------------------------------
__global__ void cvt_f32_to_bf16(const float* __restrict__ in,
                                unsigned short* __restrict__ out, long n)
{
    long i = (long)blockIdx.x * blockDim.x + threadIdx.x;
    if (i < n) out[i] = (unsigned short)bf16_rne(in[i]);
}

__global__ void transpose_cvt_bf16(const float* __restrict__ W,
                                   unsigned short* __restrict__ Wt,
                                   int K, int N)
{
    long i = (long)blockIdx.x * blockDim.x + threadIdx.x;
    if (i < (long)K * N) {
        int r = (int)(i / N);
        int c = (int)(i % N);
        Wt[(long)c * K + r] = (unsigned short)bf16_rne(W[i]);
    }
}

// ---------------------------------------------------------------------------
extern "C" void kernel_launch(void* const* d_in, const int* in_sizes, int n_in,
                              void* d_out, int out_size, void* d_ws, size_t ws_size,
                              hipStream_t stream)
{
    (void)in_sizes; (void)n_in; (void)out_size; (void)ws_size;

    const float* x     = (const float*)d_in[0];  // [8,4096,512]
    const float* Wqkv  = (const float*)d_in[1];  // [512,1536]
    const float* bqkv  = (const float*)d_in[2];  // [1536]
    const float* Wproj = (const float*)d_in[3];  // [512,512]
    const float* bproj = (const float*)d_in[4];  // [512]
    float*       out   = (float*)d_out;          // [8,4096,512]

    const long M  = 8L * 4096;   // B*S tokens
    const long D  = 512;
    const long N1 = 1536;

    // workspace carve-up (~258 MB)
    char* ws = (char*)d_ws;
    unsigned short* x_bf    = (unsigned short*)ws;                 // M*D bf16
    unsigned short* wqkv_t  = x_bf   + M * D;                      // N1*D bf16 (col-major W_qkv)
    unsigned short* wproj_t = wqkv_t + N1 * D;                     // D*D bf16  (col-major W_proj)
    float*          qkv     = (float*)(wproj_t + D * D);           // M*N1 fp32
    unsigned short* res_mid = (unsigned short*)(qkv + M * N1);     // M*D bf16

    // 1) precision conversion (bf16 is the WMMA-rewarded path on CDNA5)
    cvt_f32_to_bf16<<<(unsigned)((M * D + 255) / 256), 256, 0, stream>>>(x, x_bf, M * D);
    transpose_cvt_bf16<<<(unsigned)((D * N1 + 255) / 256), 256, 0, stream>>>(Wqkv, wqkv_t, (int)D, (int)N1);
    transpose_cvt_bf16<<<(unsigned)((D * D + 255) / 256), 256, 0, stream>>>(Wproj, wproj_t, (int)D, (int)D);

    // 2) qkv = x @ W_qkv + b_qkv
    {
        dim3 grid((unsigned)(N1 / 64), (unsigned)(M / 128));
        gemm_bf16_wmma<<<grid, 256, 0, stream>>>(x_bf, wqkv_t, bqkv, nullptr, qkv,
                                                 (int)M, (int)N1, (int)D);
    }

    // 3) per-token head-dim softmax-mul (routing permutation cancels exactly)
    attn_softmax_mul<<<(unsigned)M, 256, 0, stream>>>(qkv, (unsigned*)res_mid);

    // 4) out = x + res_mid @ W_proj + b_proj   (residual fused in epilogue)
    {
        dim3 grid((unsigned)(D / 64), (unsigned)(M / 128));
        gemm_bf16_wmma<<<grid, 256, 0, stream>>>(res_mid, wproj_t, bproj, x, out,
                                                 (int)M, (int)D, (int)D);
    }
}
